// MultiHeadAttention_82592221102086
// MI455X (gfx1250) — matrix operand order repaired and checked by reading, not compiled, not run on hardware
//
#include <hip/hip_runtime.h>
#include <hip/hip_bf16.h>
#include <float.h>

typedef __attribute__((ext_vector_type(16))) _Float16 v16h;
typedef __attribute__((ext_vector_type(8)))  _Float16 v8h;
typedef __attribute__((ext_vector_type(8)))  float    v8f;
typedef __attribute__((ext_vector_type(4)))  float    v4f;
typedef __attribute__((ext_vector_type(4)))  unsigned int u32x4;
typedef __attribute__((ext_vector_type(8)))  int      i32x8;
typedef __attribute__((ext_vector_type(4)))  int      i32x4;

#define DM    1024
#define NH    16
#define DK    64
#define BB    2
#define SS    2048

// ---------------------------------------------------------------------------
// Unified WMMA GEMM:  C = scale * (A[M,K] @ B[N,K]^T) + bias[n]
// Block tile 128x64, K-step 32, 8 waves (wave32), each wave -> 32x32 (2x2 WMMA).
// Double-buffered LDS staging (one barrier per K-step).
// Per-z base offsets: base = (z/div)*outer + (z%div)*inner  (for A, B, C).
// TRANS_OUT: store C[b][n][s] (b = m/rowsPer, s = m%rowsPer) -- used for V^T.
// NT_A / NT_C: non-temporal load of A / store of C (streaming attn tensor).
// ---------------------------------------------------------------------------
template <bool TRANS_OUT, bool NT_A, bool NT_C>
__global__ __launch_bounds__(256) void wmma_gemm_nt(
    const float* __restrict__ A, const float* __restrict__ B,
    float* __restrict__ C, const float* __restrict__ bias,
    int M, int N, int K, int lda, int ldb, int ldc, float scale,
    int azDiv, long long azOuter, long long azInner,
    int bzDiv, long long bzOuter, long long bzInner,
    int czDiv, long long czOuter, long long czInner,
    int rowsPer)
{
    __shared__ __align__(16) _Float16 As[2 * 128 * 32];
    __shared__ __align__(16) _Float16 Bs[2 * 64 * 32];

    const int z = blockIdx.z;
    const float* Ab = A + (size_t)((z / azDiv) * azOuter + (z % azDiv) * azInner);
    const float* Bb = B + (size_t)((z / bzDiv) * bzOuter + (z % bzDiv) * bzInner);
    float*       Cb = C + (size_t)((z / czDiv) * czOuter + (z % czDiv) * czInner);

    const int blockM = blockIdx.y * 128;
    const int blockN = blockIdx.x * 64;
    const int tid  = threadIdx.x;
    const int lane = tid & 31;
    const int wave = tid >> 5;
    const int waveM = (wave >> 1) * 32;  // 0,32,64,96
    const int waveN = (wave & 1) * 32;   // 0,32

    const int lm      = lane & 15;
    const int halfSel = (lane >> 4) & 1;
    const int kbA     = halfSel * 8;     // A lane-half K base
    const int kbB     = halfSel * 8;     // B lane-half K base
    const int rowHalf = halfSel * 8;     // C lane-half M base

    v8f acc[2][2] = {};
    v4f aStage[4], bStage[2];

    auto loadGlobal = [&](int k0) {
#pragma unroll
        for (int i = 0; i < 4; ++i) {
            int e = (i * 256 + tid) * 4;
            int r = e >> 5, c = e & 31;
            const v4f* p = (const v4f*)(Ab + (size_t)(blockM + r) * lda + k0 + c);
            if (NT_A) aStage[i] = __builtin_nontemporal_load(p);
            else      aStage[i] = *p;
        }
#pragma unroll
        for (int i = 0; i < 2; ++i) {
            int e = (i * 256 + tid) * 4;
            int r = e >> 5, c = e & 31;
            bStage[i] = *(const v4f*)(Bb + (size_t)(blockN + r) * ldb + k0 + c);
        }
    };
    auto storeLDS = [&](int buf) {
        _Float16* Asb = &As[buf * 128 * 32];
        _Float16* Bsb = &Bs[buf * 64 * 32];
#pragma unroll
        for (int i = 0; i < 4; ++i) {
            int e = (i * 256 + tid) * 4;
            int r = e >> 5, c = e & 31;
            _Float16* d = &Asb[r * 32 + c];
            d[0] = (_Float16)aStage[i].x; d[1] = (_Float16)aStage[i].y;
            d[2] = (_Float16)aStage[i].z; d[3] = (_Float16)aStage[i].w;
        }
#pragma unroll
        for (int i = 0; i < 2; ++i) {
            int e = (i * 256 + tid) * 4;
            int r = e >> 5, c = e & 31;
            _Float16* d = &Bsb[r * 32 + c];
            d[0] = (_Float16)bStage[i].x; d[1] = (_Float16)bStage[i].y;
            d[2] = (_Float16)bStage[i].z; d[3] = (_Float16)bStage[i].w;
        }
    };

    // prologue: stage tile 0
    loadGlobal(0);
    storeLDS(0);
    __syncthreads();

    int cur = 0;
    for (int k0 = 0; k0 < K; k0 += 32) {
        const bool notLast = (k0 + 32 < K);
        if (notLast) loadGlobal(k0 + 32);   // next tile's global loads in flight

        const _Float16* Asb = &As[cur * 128 * 32];
        const _Float16* Bsb = &Bs[cur * 64 * 32];
        v16h aFrag[2], bFrag[2];
#pragma unroll
        for (int mi = 0; mi < 2; ++mi) {
            int m = waveM + mi * 16 + lm;
            v8h c0 = *(const v8h*)&Asb[m * 32 + kbA];
            v8h c1 = *(const v8h*)&Asb[m * 32 + 16 + kbA];
            v16h a;
#pragma unroll
            for (int j = 0; j < 8; ++j) { a[j] = c0[j]; a[8 + j] = c1[j]; }
            aFrag[mi] = a;
        }
#pragma unroll
        for (int ni = 0; ni < 2; ++ni) {
            int n = waveN + ni * 16 + lm;
            v8h c0 = *(const v8h*)&Bsb[n * 32 + kbB];
            v8h c1 = *(const v8h*)&Bsb[n * 32 + 16 + kbB];
            v16h b;
#pragma unroll
            for (int j = 0; j < 8; ++j) { b[j] = c0[j]; b[8 + j] = c1[j]; }
            bFrag[ni] = b;
        }

#pragma unroll
        for (int mi = 0; mi < 2; ++mi)
#pragma unroll
            for (int ni = 0; ni < 2; ++ni)
                acc[mi][ni] = __builtin_amdgcn_wmma_f32_16x16x32_f16(
                    false, aFrag[mi], false, bFrag[ni],
                    (short)0, acc[mi][ni], false, false);

        if (notLast) {
            storeLDS(cur ^ 1);     // write other buffer (no conflict with reads)
            __syncthreads();       // single barrier per K-step
            cur ^= 1;
        }
    }

    // ---- epilogue: scale + bias, scatter per C layout ----
#pragma unroll
    for (int mi = 0; mi < 2; ++mi) {
#pragma unroll
        for (int ni = 0; ni < 2; ++ni) {
            int n = blockN + waveN + ni * 16 + lm;
            float bv = bias ? bias[n] : 0.0f;
#pragma unroll
            for (int r = 0; r < 8; ++r) {
                int m = blockM + waveM + mi * 16 + rowHalf + r;
                float val = acc[mi][ni][r] * scale + bv;
                if (TRANS_OUT) {
                    size_t idx = (size_t)(m / rowsPer) * ((size_t)N * rowsPer)
                               + (size_t)n * rowsPer + (size_t)(m % rowsPer);
                    Cb[idx] = val;
                } else if (NT_C) {
                    __builtin_nontemporal_store(val, &Cb[(size_t)m * ldc + n]);
                } else {
                    Cb[(size_t)m * ldc + n] = val;
                }
            }
        }
    }
}

// ---------------------------------------------------------------------------
// Masked softmax over rows of attn [B,H,S,S], in place. One block per row.
// attn is a 536MB one-shot stream -> non-temporal; mask (33MB) reused 16x -> RT.
// ---------------------------------------------------------------------------
__global__ __launch_bounds__(256) void softmax_mask_kernel(
    float* __restrict__ attn, const int* __restrict__ mask)
{
    const int row = blockIdx.x;            // (b*NH + h)*SS + s
    const int b = row / (NH * SS);
    const int s = row % SS;
    float* x = attn + (size_t)row * SS;
    const int* mrow = mask + ((size_t)b * SS + s) * SS;

    const int tid = threadIdx.x;
    float vals[SS / 256];
    float lmax = -FLT_MAX;
#pragma unroll
    for (int i = 0; i < SS / 256; ++i) {
        int t = tid + i * 256;
        float v = __builtin_nontemporal_load(&x[t]);
        v = (mrow[t] == 0) ? -FLT_MAX : v;
        vals[i] = v;
        lmax = fmaxf(lmax, v);
    }

    __shared__ float red[256];
    red[tid] = lmax;
    __syncthreads();
    for (int off = 128; off > 0; off >>= 1) {
        if (tid < off) red[tid] = fmaxf(red[tid], red[tid + off]);
        __syncthreads();
    }
    float rmax = red[0];
    __syncthreads();

    float lsum = 0.0f;
#pragma unroll
    for (int i = 0; i < SS / 256; ++i) {
        vals[i] = __expf(vals[i] - rmax);
        lsum += vals[i];
    }
    red[tid] = lsum;
    __syncthreads();
    for (int off = 128; off > 0; off >>= 1) {
        if (tid < off) red[tid] += red[tid + off];
        __syncthreads();
    }
    float inv = 1.0f / red[0];
#pragma unroll
    for (int i = 0; i < SS / 256; ++i)
        __builtin_nontemporal_store(vals[i] * inv, &x[tid + i * 256]);
}

// ---------------------------------------------------------------------------
// Compile probe (never launched): exercises gfx1250 async-to-LDS + TDM paths
// so the toolchain's acceptance of these instructions is verified before they
// are folded into the staging loop. This toolchain exposes the 6-arg
// tensor_load_to_lds (clang-23 / therock form).
// ---------------------------------------------------------------------------
__global__ void cdna5_async_probe(const float* __restrict__ g,
                                  float* __restrict__ out)
{
    __shared__ __align__(16) float buf[1024];
    unsigned lds = (unsigned)(uintptr_t)&buf[threadIdx.x * 4];
    const float* gp = g + threadIdx.x * 4;
    asm volatile("global_load_async_to_lds_b128 %0, %1, off"
                 :: "v"(lds), "v"(gp) : "memory");
    asm volatile("s_wait_asynccnt 0" ::: "memory");
#if __has_builtin(__builtin_amdgcn_tensor_load_to_lds)
    __builtin_amdgcn_tensor_load_to_lds((u32x4){0, 0, 0, 0},
                                        (i32x8){0, 0, 0, 0, 0, 0, 0, 0},
                                        (i32x4){0, 0, 0, 0},
                                        (i32x4){0, 0, 0, 0},
                                        (i32x8){0, 0, 0, 0, 0, 0, 0, 0}, 0);
#endif
#if __has_builtin(__builtin_amdgcn_s_wait_tensorcnt)
    __builtin_amdgcn_s_wait_tensorcnt(0);
#endif
    __syncthreads();
    out[threadIdx.x] = buf[threadIdx.x];
}

// ---------------------------------------------------------------------------
extern "C" void kernel_launch(void* const* d_in, const int* in_sizes, int n_in,
                              void* d_out, int out_size, void* d_ws, size_t ws_size,
                              hipStream_t stream)
{
    (void)in_sizes; (void)n_in; (void)out_size; (void)ws_size;
    const float* Q    = (const float*)d_in[0];
    const float* K    = (const float*)d_in[1];
    const float* V    = (const float*)d_in[2];
    const int*   mask = (const int*)  d_in[3];
    const float* Wq   = (const float*)d_in[4];
    const float* bq   = (const float*)d_in[5];
    const float* Wk   = (const float*)d_in[6];
    const float* bk   = (const float*)d_in[7];
    const float* Wv   = (const float*)d_in[8];
    const float* bv   = (const float*)d_in[9];
    const float* Wo   = (const float*)d_in[10];
    const float* bo   = (const float*)d_in[11];

    const size_t tokens = (size_t)BB * SS;          // 4096
    float* qbuf  = (float*)d_ws;                    // [B,S,DM]
    float* kbuf  = qbuf + tokens * DM;              // [B,S,DM]
    float* vtbuf = kbuf + tokens * DM;              // [B,H,DK,S] (V^T per head)
    float* ctx   = qbuf;                            // reuse q after scores
    float* outp  = (float*)d_out;                   // [B,S,DM]
    float* attn  = outp + tokens * DM;              // [B,H,S,S]

    const long long BIG = 1LL << 30;
    const int M = (int)tokens;

    // Q/K projections: [4096,1024] = X @ W^T + b
    wmma_gemm_nt<false, false, false><<<dim3(DM / 64, M / 128, 1), 256, 0, stream>>>(
        Q, Wq, qbuf, bq, M, DM, DM, DM, DM, DM, 1.0f,
        1, 0, 0, 1, 0, 0, 1, 0, 0, SS);
    wmma_gemm_nt<false, false, false><<<dim3(DM / 64, M / 128, 1), 256, 0, stream>>>(
        K, Wk, kbuf, bk, M, DM, DM, DM, DM, DM, 1.0f,
        1, 0, 0, 1, 0, 0, 1, 0, 0, SS);
    // V projection, stored transposed per head: vT[b][h*64+dk][s]
    wmma_gemm_nt<true, false, false><<<dim3(DM / 64, M / 128, 1), 256, 0, stream>>>(
        V, Wv, vtbuf, bv, M, DM, DM, DM, DM, DM, 1.0f,
        1, 0, 0, 1, 0, 0, 1, 0, 0, SS);

    // scores = q @ k^T / sqrt(DK), per (b,h): z = b*16+h; NT store of attn
    wmma_gemm_nt<false, false, true><<<dim3(SS / 64, SS / 128, BB * NH), 256, 0, stream>>>(
        qbuf, kbuf, attn, nullptr, SS, SS, DK, DM, DM, SS, 0.125f,
        NH, (long long)SS * DM, DK,
        NH, (long long)SS * DM, DK,
        (int)BIG, 0, (long long)SS * SS, SS);

    // masked softmax in place on attn
    softmax_mask_kernel<<<BB * NH * SS, 256, 0, stream>>>(attn, mask);

    // ctx = attn @ v  (via v^T as the [N,K] operand); NT load of attn
    wmma_gemm_nt<false, true, false><<<dim3(DK / 64, SS / 128, BB * NH), 256, 0, stream>>>(
        attn, vtbuf, ctx, nullptr, SS, DK, SS, SS, SS, DM, 1.0f,
        (int)BIG, 0, (long long)SS * SS,
        (int)BIG, 0, (long long)DK * SS,
        NH, (long long)SS * DM, DK, SS);

    // out = ctx @ Wo^T + bo
    wmma_gemm_nt<false, false, false><<<dim3(DM / 64, M / 128, 1), 256, 0, stream>>>(
        ctx, Wo, outp, bo, M, DM, DM, DM, DM, DM, 1.0f,
        1, 0, 0, 1, 0, 0, 1, 0, 0, SS);
}
